// S5_SSM_5222680231981
// MI455X (gfx1250) — matrix operand order repaired and checked by reading, not compiled, not run on hardware
//
#include <hip/hip_runtime.h>
#include <hip/hip_bf16.h>

typedef __bf16 bf16_t;
typedef bf16_t v16bf __attribute__((ext_vector_type(16)));
typedef float  v8f   __attribute__((ext_vector_type(8)));

#define B_SZ   8
#define LSEQ   8192
#define H_DIM  1024
#define P_DIM  64
#define NROWS  (B_SZ * LSEQ)      // 65536 total (b,l) rows
#define KQ     128                // 2*P: [real | imag] channels
#define TCHUNK 128                // scan chunk length
#define NCHUNK (LSEQ / TCHUNK)    // 64 chunks per batch

// ---------------------------------------------------------------------------
// K0: per-channel scalars: dt = softplus(inv_dt), bilinear dA, dt/den,
//     and dA^(t+1) for t = 0..TCHUNK-1 (carry-adjustment powers).
// ---------------------------------------------------------------------------
__global__ void k_setup_scalars(const float* __restrict__ A_real,
                                const float* __restrict__ A_imag,
                                const float* __restrict__ inv_dt,
                                float* __restrict__ dA_r, float* __restrict__ dA_i,
                                float* __restrict__ dBfac_r, float* __restrict__ dBfac_i,
                                float* __restrict__ dApow_r, float* __restrict__ dApow_i) {
  int p = threadIdx.x;
  if (p >= P_DIM) return;
  float x  = inv_dt[p];
  float dt = (x > 20.0f) ? x : log1pf(__expf(x));
  float hr = 0.5f * dt * A_real[p];
  float hi = 0.5f * dt * A_imag[p];
  float den_r = 1.0f - hr, den_i = -hi;
  float inv   = 1.0f / (den_r * den_r + den_i * den_i);
  float num_r = 1.0f + hr, num_i = hi;
  float dar = (num_r * den_r + num_i * den_i) * inv;   // (1+dtA/2)/(1-dtA/2)
  float dai = (num_i * den_r - num_r * den_i) * inv;
  dA_r[p] = dar; dA_i[p] = dai;
  dBfac_r[p] =  dt * den_r * inv;                       // dt / den
  dBfac_i[p] = -dt * den_i * inv;
  float pr = dar, pi = dai;                             // dA^1 ... dA^T
  for (int t = 0; t < TCHUNK; ++t) {
    dApow_r[t * P_DIM + p] = pr;
    dApow_i[t * P_DIM + p] = pi;
    float nr = pr * dar - pi * dai;
    float ni = pr * dai + pi * dar;
    pr = nr; pi = ni;
  }
}

// ---------------------------------------------------------------------------
// K1: build bf16 weight matrices (stay L2-resident, 256 KB each):
//   dBbf [q=0..127][h]: q<64 -> Re((dt/den)*B)[q,h], q>=64 -> Im(...)[q-64,h]
//   Ccomb[h][k=0..127]: k<64 -> C_real[h,k], k>=64 -> -C_imag[h,k-64]
// ---------------------------------------------------------------------------
__global__ void k_setup_convert(const float* __restrict__ B_real,
                                const float* __restrict__ B_imag,
                                const float* __restrict__ C_real,
                                const float* __restrict__ C_imag,
                                const float* __restrict__ dBfac_r,
                                const float* __restrict__ dBfac_i,
                                bf16_t* __restrict__ dBbf,
                                bf16_t* __restrict__ Ccomb) {
  int idx = blockIdx.x * blockDim.x + threadIdx.x;
  if (idx < KQ * H_DIM) {
    int q = idx >> 10, h = idx & (H_DIM - 1);
    int p = q & (P_DIM - 1);
    float fr = dBfac_r[p], fi = dBfac_i[p];
    float br = B_real[p * H_DIM + h], bi = B_imag[p * H_DIM + h];
    float v  = (q < P_DIM) ? (fr * br - fi * bi) : (fr * bi + fi * br);
    dBbf[idx] = (bf16_t)v;
  } else if (idx < 2 * KQ * H_DIM) {
    int j = idx - KQ * H_DIM;
    int h = j >> 7, k = j & (KQ - 1);
    float v = (k < P_DIM) ? C_real[h * P_DIM + k] : -C_imag[h * P_DIM + (k - P_DIM)];
    Ccomb[j] = (bf16_t)v;
  }
}

// ---------------------------------------------------------------------------
// K2: input GEMM  Bu[n, q] = sum_h u[n,h] * dBbf[q,h]   (n rows, q = 2P cols)
// 512 blocks x 8 waves; block tile 128x128, wave tile 16x128, K step 32.
// All 8 B fragments for a K-step are batch-issued (one load clause) before
// the 8 independent WMMAs, so only one wait precedes the WMMA burst.
// ---------------------------------------------------------------------------
__global__ __launch_bounds__(256) void k_gemm1(const float* __restrict__ u,
                                               const bf16_t* __restrict__ dBbf,
                                               float* __restrict__ Bu_r,
                                               float* __restrict__ Bu_i) {
  const int wave = threadIdx.x >> 5;
  const int lane = threadIdx.x & 31;
  const int m  = lane & 15;
  const int hi = lane >> 4;
  const int row = blockIdx.x * 128 + wave * 16 + m;   // A-fragment row (M)

  v8f acc[8];
  #pragma unroll
  for (int c = 0; c < 8; ++c) acc[c] = (v8f){0.f, 0.f, 0.f, 0.f, 0.f, 0.f, 0.f, 0.f};

  for (int k0 = 0; k0 < H_DIM; k0 += 32) {
    // Batch-issue all 8 B fragments (16x b128 loads, L2-resident weights).
    v16bf bfr[8];
    #pragma unroll
    for (int c = 0; c < 8; ++c) {
      int q = c * 16 + m;                      // B column (striped over lanes 0..15)
      bfr[c] = *(const v16bf*)(dBbf + (size_t)q * H_DIM + k0 + hi * 16);
    }
    // A fragment: lane holds K = hi*16+{0..15} of its row;
    // loads + f32->bf16 conversion overlap with the B loads in flight.
    v16bf a;
    const float* up0 = u + (size_t)row * H_DIM + k0 + hi * 16;
    const float* up1 = up0 + 8;
    #pragma unroll
    for (int i = 0; i < 8; ++i) { a[i] = (bf16_t)up0[i]; a[8 + i] = (bf16_t)up1[i]; }
    #pragma unroll
    for (int c = 0; c < 8; ++c) {
      acc[c] = __builtin_amdgcn_wmma_f32_16x16x32_bf16(false, a, false, bfr[c],
                                                       (short)0, acc[c], false, false);
    }
  }
  #pragma unroll
  for (int c = 0; c < 8; ++c) {
    int q = c * 16 + m;
    float* dst = (q < P_DIM) ? (Bu_r + q) : (Bu_i + (q - P_DIM));
    #pragma unroll
    for (int j = 0; j < 8; ++j) {
      int n = blockIdx.x * 128 + wave * 16 + j + 8 * hi;  // D element j -> M = j+8*hi
      dst[(size_t)n * P_DIM] = acc[c][j];
    }
  }
}

// ---------------------------------------------------------------------------
// K3: per-chunk local scans (carry-in 0, except chunk 0 which seeds h0).
// One thread per (b, chunk, p); in-place Bu -> h_local; writes chunk sums S.
// ---------------------------------------------------------------------------
__global__ void k_scan1(const float* __restrict__ dA_r, const float* __restrict__ dA_i,
                        const float* __restrict__ h0_r, const float* __restrict__ h0_i,
                        float* __restrict__ Bu_r, float* __restrict__ Bu_i,
                        float* __restrict__ S_r, float* __restrict__ S_i) {
  int tid   = blockIdx.x * blockDim.x + threadIdx.x;   // 32768 threads
  int p     = tid & (P_DIM - 1);
  int chunk = (tid >> 6) & (NCHUNK - 1);
  int b     = tid >> 12;
  float dar = dA_r[p], dai = dA_i[p];
  float hr = 0.f, hi = 0.f;
  if (chunk == 0) { hr = h0_r[b * P_DIM + p]; hi = h0_i[b * P_DIM + p]; }
  size_t base = ((size_t)b * LSEQ + (size_t)chunk * TCHUNK) * P_DIM + p;
  for (int t = 0; t < TCHUNK; ++t) {
    size_t idx = base + (size_t)t * P_DIM;
    float br = Bu_r[idx], bi = Bu_i[idx];
    float nr = dar * hr - dai * hi + br;
    float ni = dar * hi + dai * hr + bi;
    hr = nr; hi = ni;
    Bu_r[idx] = hr; Bu_i[idx] = hi;
  }
  int si = (b * NCHUNK + chunk) * P_DIM + p;
  S_r[si] = hr; S_i[si] = hi;
}

// ---------------------------------------------------------------------------
// K4: chunk-prefix per (b,p): carry_in[c] = dA^T * carry_in[c-1] + S[c-1];
// final carry is h_last -> written to output tail.
// ---------------------------------------------------------------------------
__global__ void k_scan2(const float* __restrict__ dA_r, const float* __restrict__ dA_i,
                        const float* __restrict__ S_r, const float* __restrict__ S_i,
                        float* __restrict__ carry_r, float* __restrict__ carry_i,
                        float* __restrict__ out_hlast) {
  int tid = blockIdx.x * blockDim.x + threadIdx.x;     // 512 threads
  int p = tid & (P_DIM - 1);
  int b = tid >> 6;
  float tr = dA_r[p], ti = dA_i[p];                    // dA^128 via 7 squarings
  for (int s = 0; s < 7; ++s) { float nr = tr * tr - ti * ti; ti = 2.f * tr * ti; tr = nr; }
  float cr = 0.f, ci = 0.f;
  for (int c = 0; c < NCHUNK; ++c) {
    int si = (b * NCHUNK + c) * P_DIM + p;
    carry_r[si] = cr; carry_i[si] = ci;
    float sr = S_r[si], sii = S_i[si];
    float nr = tr * cr - ti * ci + sr;
    float ni = tr * ci + ti * cr + sii;
    cr = nr; ci = ni;
  }
  out_hlast[b * P_DIM + p] = cr;                       // h_last.real
  out_hlast[B_SZ * P_DIM + b * P_DIM + p] = ci;        // h_last.imag
}

// ---------------------------------------------------------------------------
// K5: output GEMM. Per (b,chunk) block: rebuild hs tile (128 t x 128 q) in LDS
// as bf16 (hs = h_local + dA^(t+1)*carry), then y = 2 * hs @ [C_r;-C_i] + D*u.
// ---------------------------------------------------------------------------
__global__ __launch_bounds__(256) void k_gemm2(
    const float* __restrict__ Hloc_r, const float* __restrict__ Hloc_i,
    const float* __restrict__ carry_r, const float* __restrict__ carry_i,
    const float* __restrict__ dApow_r, const float* __restrict__ dApow_i,
    const bf16_t* __restrict__ Ccomb,
    const float* __restrict__ u, const float* __restrict__ D,
    float* __restrict__ y) {
  __shared__ bf16_t hs[TCHUNK * 130];                  // +2 pad vs 128
  const int n0    = blockIdx.x * TCHUNK;
  const int b     = n0 / LSEQ;
  const int chunk = (n0 % LSEQ) / TCHUNK;
  const int cbase = (b * NCHUNK + chunk) * P_DIM;

  for (int idx = threadIdx.x; idx < TCHUNK * KQ; idx += 256) {
    int t = idx >> 7, q = idx & (KQ - 1);
    int p = q & (P_DIM - 1);
    float cr = carry_r[cbase + p], ci = carry_i[cbase + p];
    float pr = dApow_r[t * P_DIM + p], pi = dApow_i[t * P_DIM + p];
    size_t hidx = ((size_t)(n0 + t)) * P_DIM + p;
    float v = (q < P_DIM) ? (Hloc_r[hidx] + (pr * cr - pi * ci))
                          : (Hloc_i[hidx] + (pr * ci + pi * cr));
    hs[t * 130 + q] = (bf16_t)v;
  }
  __syncthreads();

  const int wave = threadIdx.x >> 5;
  const int lane = threadIdx.x & 31;
  const int m  = lane & 15;
  const int hi = lane >> 4;

  v16bf afr[4];                                        // A frags for all 4 K-steps
  #pragma unroll
  for (int ks = 0; ks < 4; ++ks) {
    int krow = (wave * 16 + m) * 130;
    int k0 = ks * 32;
    #pragma unroll
    for (int i = 0; i < 8; ++i) {
      afr[ks][i]     = hs[krow + k0 + hi * 16 + i];
      afr[ks][8 + i] = hs[krow + k0 + hi * 16 + 8 + i];
    }
  }

  for (int c = 0; c < H_DIM / 16; ++c) {
    int hcol = c * 16 + m;
    // Batch-issue the 4 B fragments for this column tile, then chain WMMAs.
    v16bf bfr[4];
    #pragma unroll
    for (int ks = 0; ks < 4; ++ks)
      bfr[ks] = *(const v16bf*)(Ccomb + (size_t)hcol * KQ + ks * 32 + hi * 16);
    v8f acc = (v8f){0.f, 0.f, 0.f, 0.f, 0.f, 0.f, 0.f, 0.f};
    #pragma unroll
    for (int ks = 0; ks < 4; ++ks) {
      acc = __builtin_amdgcn_wmma_f32_16x16x32_bf16(false, afr[ks], false, bfr[ks],
                                                    (short)0, acc, false, false);
    }
    float dv = D[hcol];
    #pragma unroll
    for (int j = 0; j < 8; ++j) {
      int n = n0 + wave * 16 + j + 8 * hi;
      size_t oi = (size_t)n * H_DIM + hcol;
      y[oi] = 2.f * acc[j] + dv * u[oi];
    }
  }
}

// ---------------------------------------------------------------------------
extern "C" void kernel_launch(void* const* d_in, const int* in_sizes, int n_in,
                              void* d_out, int out_size, void* d_ws, size_t ws_size,
                              hipStream_t stream) {
  const float* u      = (const float*)d_in[0];
  const float* h0_r   = (const float*)d_in[1];
  const float* h0_i   = (const float*)d_in[2];
  const float* A_r    = (const float*)d_in[3];
  const float* A_i    = (const float*)d_in[4];
  const float* B_r    = (const float*)d_in[5];
  const float* B_i    = (const float*)d_in[6];
  const float* C_r    = (const float*)d_in[7];
  const float* C_i    = (const float*)d_in[8];
  const float* Dv     = (const float*)d_in[9];
  const float* inv_dt = (const float*)d_in[10];
  float* y = (float*)d_out;
  float* out_hlast = y + (size_t)NROWS * H_DIM;

  char* ws = (char*)d_ws;
  size_t off = 0;
  auto alloc = [&](size_t bytes) -> char* {
    char* p = ws + off;
    off += (bytes + 255) & ~(size_t)255;
    return p;
  };
  float*  Bu_r    = (float*)alloc((size_t)NROWS * P_DIM * 4);   // 16 MB (-> h_local)
  float*  Bu_i    = (float*)alloc((size_t)NROWS * P_DIM * 4);   // 16 MB
  float*  dA_r    = (float*)alloc(P_DIM * 4);
  float*  dA_i    = (float*)alloc(P_DIM * 4);
  float*  dBfac_r = (float*)alloc(P_DIM * 4);
  float*  dBfac_i = (float*)alloc(P_DIM * 4);
  float*  dApow_r = (float*)alloc(TCHUNK * P_DIM * 4);
  float*  dApow_i = (float*)alloc(TCHUNK * P_DIM * 4);
  float*  S_r     = (float*)alloc(B_SZ * NCHUNK * P_DIM * 4);
  float*  S_i     = (float*)alloc(B_SZ * NCHUNK * P_DIM * 4);
  float*  carry_r = (float*)alloc(B_SZ * NCHUNK * P_DIM * 4);
  float*  carry_i = (float*)alloc(B_SZ * NCHUNK * P_DIM * 4);
  bf16_t* dBbf    = (bf16_t*)alloc((size_t)KQ * H_DIM * 2);     // 256 KB
  bf16_t* Ccomb   = (bf16_t*)alloc((size_t)H_DIM * KQ * 2);     // 256 KB

  k_setup_scalars<<<1, 64, 0, stream>>>(A_r, A_i, inv_dt, dA_r, dA_i,
                                        dBfac_r, dBfac_i, dApow_r, dApow_i);
  k_setup_convert<<<(2 * KQ * H_DIM) / 256, 256, 0, stream>>>(
      B_r, B_i, C_r, C_i, dBfac_r, dBfac_i, dBbf, Ccomb);
  k_gemm1<<<NROWS / 128, 256, 0, stream>>>(u, dBbf, Bu_r, Bu_i);
  k_scan1<<<(B_SZ * NCHUNK * P_DIM) / 256, 256, 0, stream>>>(
      dA_r, dA_i, h0_r, h0_i, Bu_r, Bu_i, S_r, S_i);
  k_scan2<<<(B_SZ * P_DIM) / 256, 256, 0, stream>>>(
      dA_r, dA_i, S_r, S_i, carry_r, carry_i, out_hlast);
  k_gemm2<<<NROWS / TCHUNK, 256, 0, stream>>>(
      Bu_r, Bu_i, carry_r, carry_i, dApow_r, dApow_i, Ccomb, u, Dv, y);
  (void)in_sizes; (void)n_in; (void)out_size; (void)ws_size;
}
